// Local_Module_81363860455643
// MI455X (gfx1250) — matrix operand order repaired and checked by reading, not compiled, not run on hardware
//
#include <hip/hip_runtime.h>
#include <cstdint>

// ---------------------------------------------------------------------------
// Types for CDNA5 WMMA (wave32): v16bf A/B fragments, v8f f32 accumulator.
// ---------------------------------------------------------------------------
typedef __attribute__((ext_vector_type(16))) __bf16 v16bf;
typedef __attribute__((ext_vector_type(8)))  float  v8f;

struct U128x2 { uint4 lo, hi; };

static __device__ __forceinline__ v16bf pack_v16bf(uint4 lo, uint4 hi) {
    U128x2 t; t.lo = lo; t.hi = hi;
    return __builtin_bit_cast(v16bf, t);
}

static __device__ __forceinline__ unsigned short f2bf(float f) {
    unsigned int u = __builtin_bit_cast(unsigned int, f);
    u += 0x7fffu + ((u >> 16) & 1u);          // round-to-nearest-even
    return (unsigned short)(u >> 16);
}

#define CONST_C    1024
#define CONST_C4   256
#define CONST_HW   9216   // 96*96
#define CONST_H    96
#define CONST_W    96

// ---------------------------------------------------------------------------
// Kernel 1: 1x1 conv as GEMM  D[o,hw] = sum_c Wt[o,c]*X[c,hw] + bias[o]
// Output stored TRANSPOSED in bf16: outT[b][hw][o]  (channel-contiguous),
// so the correlation kernel can load WMMA fragments directly from global.
// Workgroup tile 128(M) x 128(N); 8 waves, each 64x32 (4x2 WMMA tiles).
// Double-buffered LDS staging: next K-tile's global loads overlap WMMAs.
// ---------------------------------------------------------------------------
__global__ __launch_bounds__(256)
void conv1x1_bf16T(const float* __restrict__ X,     // [B,1024,9216] (batch via blockIdx.z)
                   const float* __restrict__ Wt,    // [256,1024]
                   const float* __restrict__ bias,  // [256]
                   unsigned short* __restrict__ outT) // [B,9216,256] bf16
{
    __shared__ __align__(16) unsigned short Wa[2][128 * 32]; // A tile [m][c], stride 32
    __shared__ __align__(16) unsigned short Xb[2][128 * 40]; // B tile [n][c], stride 40 (padded)

    const int tid  = threadIdx.x;
    const int lane = tid & 31;
    const int wv   = tid >> 5;       // 0..7
    const int wm   = wv >> 2;        // 0..1  (M half)
    const int wn   = wv & 3;         // 0..3  (N quarter)
    const int l15  = lane & 15;
    const int hi   = lane >> 4;      // 0/1 lane half

    const int nbase = blockIdx.x * 128;
    const int mbase = blockIdx.y * 128;
    const int b     = blockIdx.z;

    const float* Xg = X + (size_t)b * CONST_C * CONST_HW;

    // Per-thread staging coordinates (constant across K-steps).
    const int sa_m = (tid * 4) >> 5,  sa_c = (tid * 4) & 31;   // A: +32 m per pass
    const int sb_c = (tid * 4) >> 7,  sb_n = (tid * 4) & 127;  // B: +8 c per pass

    auto stage = [&](int kk, int buf) {
        const int c0 = kk * 32;
#pragma unroll
        for (int pass = 0; pass < 4; ++pass) {
            int m = sa_m + pass * 32;
            float4 w4 = *(const float4*)(Wt + (size_t)(mbase + m) * CONST_C + c0 + sa_c);
            uint2 p = make_uint2((unsigned)f2bf(w4.x) | ((unsigned)f2bf(w4.y) << 16),
                                 (unsigned)f2bf(w4.z) | ((unsigned)f2bf(w4.w) << 16));
            *(uint2*)&Wa[buf][m * 32 + sa_c] = p;
        }
#pragma unroll
        for (int pass = 0; pass < 4; ++pass) {
            int c = sb_c + pass * 8;
            float4 x4 = *(const float4*)(Xg + (size_t)(c0 + c) * CONST_HW + nbase + sb_n);
            Xb[buf][(sb_n + 0) * 40 + c] = f2bf(x4.x);
            Xb[buf][(sb_n + 1) * 40 + c] = f2bf(x4.y);
            Xb[buf][(sb_n + 2) * 40 + c] = f2bf(x4.z);
            Xb[buf][(sb_n + 3) * 40 + c] = f2bf(x4.w);
        }
    };

    v8f acc[4][2];
#pragma unroll
    for (int a = 0; a < 4; ++a)
#pragma unroll
        for (int bi = 0; bi < 2; ++bi)
            acc[a][bi] = v8f{0.f, 0.f, 0.f, 0.f, 0.f, 0.f, 0.f, 0.f};

    stage(0, 0);

    for (int kk = 0; kk < 32; ++kk) {
        const int cur = kk & 1;
        __syncthreads();                      // buf[cur] ready; buf[cur^1] free
        if (kk < 31) stage(kk + 1, cur ^ 1);  // overlap next tile with compute

        // ---- B fragments (per lane: N = l&15, K = 8*hi+0..7 and 16+8*hi+0..7)
        v16bf Bf[2];
#pragma unroll
        for (int bi = 0; bi < 2; ++bi) {
            int n = wn * 32 + bi * 16 + l15;
            const unsigned short* p = &Xb[cur][n * 40 + (hi ? 8 : 0)];
            Bf[bi] = pack_v16bf(*(const uint4*)p, *(const uint4*)(p + 16));
        }
        // ---- A fragments + WMMA
#pragma unroll
        for (int a = 0; a < 4; ++a) {
            int m = wm * 64 + a * 16 + l15;
            const unsigned short* p = &Wa[cur][m * 32 + (hi ? 8 : 0)];
            v16bf Af = pack_v16bf(*(const uint4*)p, *(const uint4*)(p + 16));
#pragma unroll
            for (int bi = 0; bi < 2; ++bi)
                acc[a][bi] = __builtin_amdgcn_wmma_f32_16x16x32_bf16(
                    false, Af, false, Bf[bi], (short)0, acc[a][bi], false, false);
        }
    }

    // ---- epilogue: bias + bf16 + transposed b128 store (8 consecutive o per lane)
#pragma unroll
    for (int a = 0; a < 4; ++a) {
        int o0 = mbase + wm * 64 + a * 16 + hi * 8;
        float4 ba = *(const float4*)(bias + o0);
        float4 bb = *(const float4*)(bias + o0 + 4);
        float bv[8] = {ba.x, ba.y, ba.z, ba.w, bb.x, bb.y, bb.z, bb.w};
#pragma unroll
        for (int bi = 0; bi < 2; ++bi) {
            int hw = nbase + wn * 32 + bi * 16 + l15;
            unsigned short e[8];
#pragma unroll
            for (int r = 0; r < 8; ++r) e[r] = f2bf(acc[a][bi][r] + bv[r]);
            uint4 o4;
            o4.x = (unsigned)e[0] | ((unsigned)e[1] << 16);
            o4.y = (unsigned)e[2] | ((unsigned)e[3] << 16);
            o4.z = (unsigned)e[4] | ((unsigned)e[5] << 16);
            o4.w = (unsigned)e[6] | ((unsigned)e[7] << 16);
            *(uint4*)(outT + ((size_t)b * CONST_HW + hw) * CONST_C4 + o0) = o4;
        }
    }
}

// ---------------------------------------------------------------------------
// Kernel 2: local correlation (81 displacements, zero-padded) + softmax.
// One WG = (b, h, 16-pixel w-tile). 9 waves; wave v handles h' = h-4+v and
// computes the 16x48 score band via 3 N-tiles x 8 K-steps of bf16 WMMA,
// then scatters its 9 dj values per pixel into LDS. Softmax over 81.
// ---------------------------------------------------------------------------
__global__ __launch_bounds__(288)
void corr_softmax(const unsigned short* __restrict__ qT,  // [B,9216,256] bf16
                  const unsigned short* __restrict__ kT,  // [B,9216,256] bf16
                  float* __restrict__ out)                // [B,96,96,81] f32
{
    __shared__ float corrS[16][84];

    const int i    = blockIdx.x;       // w-tile 0..5
    const int h    = blockIdx.y;       // 0..95
    const int b    = blockIdx.z;
    const int tid  = threadIdx.x;
    const int lane = tid & 31;
    const int wv   = tid >> 5;         // 0..8 -> displaced row
    const int l15  = lane & 15;
    const int hi   = lane >> 4;
    const int w0   = i * 16;
    const int hp   = h - 4 + wv;
    const bool hvalid = (hp >= 0) && (hp < CONST_H);

    v8f acc[3];
#pragma unroll
    for (int n = 0; n < 3; ++n) acc[n] = v8f{0.f, 0.f, 0.f, 0.f, 0.f, 0.f, 0.f, 0.f};

    // A fragment source: q pixels (row h, w = w0 + (l&15)), contiguous channels.
    const size_t abase =
        ((size_t)b * CONST_HW + (size_t)h * CONST_W + w0 + l15) * CONST_C4 + (hi ? 8 : 0);

    // B fragment sources: k pixels (row h', w' = w0-16+16n+(l&15)); zero if OOB.
    bool   nvld[3];
    size_t kbase[3];
#pragma unroll
    for (int n = 0; n < 3; ++n) {
        int wp = w0 - 16 + n * 16 + l15;
        bool v = hvalid && (wp >= 0) && (wp < CONST_W);
        nvld[n]  = v;
        kbase[n] = v ? (((size_t)b * CONST_HW + (size_t)hp * CONST_W + wp) * CONST_C4
                        + (hi ? 8 : 0))
                     : 0;
    }

#pragma unroll
    for (int kk = 0; kk < 8; ++kk) {
        const int c0 = kk * 32;
        const unsigned short* ap = qT + abase + c0;
        v16bf A = pack_v16bf(*(const uint4*)ap, *(const uint4*)(ap + 16));
#pragma unroll
        for (int n = 0; n < 3; ++n) {
            uint4 blo = make_uint4(0, 0, 0, 0), bhi = make_uint4(0, 0, 0, 0);
            if (nvld[n]) {
                const unsigned short* bp = kT + kbase[n] + c0;
                blo = *(const uint4*)bp;
                bhi = *(const uint4*)(bp + 16);
            }
            v16bf B = pack_v16bf(blo, bhi);
            acc[n] = __builtin_amdgcn_wmma_f32_16x16x32_bf16(
                false, A, false, B, (short)0, acc[n], false, false);
        }
    }

    // ---- band extraction: D[m=w_loc, n=w'_slab] -> dj = w'_slab - 16 - w_loc + 4
    const float inv = 1.0f / 256.0f;
#pragma unroll
    for (int n = 0; n < 3; ++n)
#pragma unroll
        for (int r = 0; r < 8; ++r) {
            int wl = r + hi * 8;
            int dj = n * 16 + l15 - 16 - wl + 4;
            if (dj >= 0 && dj < 9)
                corrS[wl][wv * 9 + dj] = acc[n][r] * inv;
        }
    __syncthreads();

    // ---- softmax over 81 per pixel, write [b,h,w,81]
    if (tid < 16) {
        const int p = tid;
        float m = corrS[p][0];
        for (int d = 1; d < 81; ++d) m = fmaxf(m, corrS[p][d]);
        float s = 0.f;
        for (int d = 0; d < 81; ++d) {
            float e = __expf(corrS[p][d] - m);
            corrS[p][d] = e;
            s += e;
        }
        float rs = 1.0f / s;
        float* op = out + (((size_t)b * CONST_H + h) * CONST_W + (w0 + p)) * 81;
        for (int d = 0; d < 81; ++d) op[d] = corrS[p][d] * rs;
    }
}

// ---------------------------------------------------------------------------
extern "C" void kernel_launch(void* const* d_in, const int* in_sizes, int n_in,
                              void* d_out, int out_size, void* d_ws, size_t ws_size,
                              hipStream_t stream) {
    (void)in_sizes; (void)n_in; (void)out_size; (void)ws_size;
    const float* x  = (const float*)d_in[0];
    const float* y  = (const float*)d_in[1];
    const float* qw = (const float*)d_in[2];
    const float* qb = (const float*)d_in[3];
    const float* kw = (const float*)d_in[4];
    const float* kb = (const float*)d_in[5];
    float* out = (float*)d_out;

    unsigned short* qT = (unsigned short*)d_ws;                       // [4,9216,256] bf16
    unsigned short* kT = qT + (size_t)4 * CONST_HW * CONST_C4;        // [4,9216,256] bf16

    dim3 g1(72, 2, 4), b1(256);
    conv1x1_bf16T<<<g1, b1, 0, stream>>>(y, qw, qb, qT);   // q = conv(y, query)
    conv1x1_bf16T<<<g1, b1, 0, stream>>>(x, kw, kb, kT);   // k = conv(x, key)

    dim3 g2(6, 96, 4), b2(288);
    corr_softmax<<<g2, b2, 0, stream>>>(qT, kT, out);
}
